// TinyAttention_79276506349629
// MI455X (gfx1250) — matrix operand order repaired and checked by reading, not compiled, not run on hardware
//
#include <hip/hip_runtime.h>
#include <hip/hip_bf16.h>

// ---------------------------------------------------------------------------
// TinyAttention for MI455X (gfx1250, wave32, WMMA)
//   DIM=1024, HEADS=16, HEAD_DIM=64, B=4, T=2048
//   bf16 WMMA (v_wmma_f32_16x16x32_bf16) for all GEMMs, fp32 accumulate.
//   f32->bf16 conversion done ONCE in a memory-bound pre-pass; GEMM tile
//   fills are pure GLOBAL_LOAD_ASYNC_TO_LDS_B128 DMA (ASYNCcnt path).
// ---------------------------------------------------------------------------

typedef __attribute__((ext_vector_type(16))) __bf16 v16bf;
typedef __attribute__((ext_vector_type(8)))  float  v8f;
typedef __attribute__((ext_vector_type(4)))  int    v4i;

union Frag16 { v16bf v; unsigned int u[8]; };

// pack two fp32 -> two bf16 in one dword (elem a in [15:0], b in [31:16])
__device__ __forceinline__ unsigned int f32x2_to_bf16x2(float a, float b) {
    unsigned ua = __float_as_uint(a), ub = __float_as_uint(b);
    unsigned ra = ua + 0x7FFFu + ((ua >> 16) & 1u);     // round-to-nearest-even
    unsigned rb = ub + 0x7FFFu + ((ub >> 16) & 1u);
    return (ra >> 16) | (rb & 0xFFFF0000u);
}

__device__ __forceinline__ unsigned short f32_to_bf16(float f) {
    unsigned u = __float_as_uint(f);
    unsigned r = u + 0x7FFFu + ((u >> 16) & 1u);
    return (unsigned short)(r >> 16);
}

#if __has_builtin(__builtin_amdgcn_global_load_async_to_lds_b128)
#define HAVE_ASYNC_LDS 1
#define ASYNC_B128(gsrc, ldst)                                               \
    __builtin_amdgcn_global_load_async_to_lds_b128(                          \
        (__attribute__((address_space(1))) v4i*)(gsrc),                      \
        (__attribute__((address_space(3))) v4i*)(ldst), 0, 0)
#else
#define HAVE_ASYNC_LDS 0
#endif

__device__ __forceinline__ void wait_async_lds() {
#if HAVE_ASYNC_LDS
#if __has_builtin(__builtin_amdgcn_s_wait_asynccnt)
    __builtin_amdgcn_s_wait_asynccnt(0);
#else
    asm volatile("s_wait_asynccnt 0" ::: "memory");
#endif
#endif
}

#define DIMC   1024
#define SEQT   2048
#define NBH    64          // B*H = 4*16
#define HD     64          // head dim

// ===========================================================================
// Kernel 0: one-shot f32 -> bf16 conversion of x and all four weights.
//   y==0: x (8M elems) ; y==1..4: Wq,Wk,Wv,Wo (1M elems each)
// ===========================================================================
__global__ __launch_bounds__(256) void convert_kernel(
    const float* __restrict__ x,
    const float* __restrict__ Wq, const float* __restrict__ Wk,
    const float* __restrict__ Wv, const float* __restrict__ Wo,
    unsigned short* __restrict__ xbf, unsigned short* __restrict__ wbf)
{
    const int y = blockIdx.y;
    const float* src;
    unsigned short* dst;
    size_t n;
    if (y == 0) {
        src = x; dst = xbf; n = (size_t)SEQT * 4 * DIMC;          // 8M
    } else {
        src = (y == 1) ? Wq : (y == 2) ? Wk : (y == 3) ? Wv : Wo;
        dst = wbf + (size_t)(y - 1) * DIMC * DIMC;
        n = (size_t)DIMC * DIMC;                                  // 1M
    }
    size_t base = ((size_t)blockIdx.x * 256 + threadIdx.x) * 8;
    if (base >= n) return;
    #pragma unroll
    for (int p = 0; p < 4; ++p) {
        float a = src[base + p * 2], b = src[base + p * 2 + 1];
        *reinterpret_cast<unsigned int*>(&dst[base + p * 2]) =
            f32x2_to_bf16x2(a, b);
    }
}

// ===========================================================================
// Kernel 1: fused QKV projection.  y = xbf @ Wbf^T, stored bf16 [bh][t][d].
// Block = 256 thr (8 waves). Tile 64(m) x 64(n), K step 32. Fills via DMA.
// ===========================================================================
__global__ __launch_bounds__(256) void qkv_proj_kernel(
    const unsigned short* __restrict__ xbf,
    const unsigned short* __restrict__ wbf,
    unsigned short* __restrict__ qkv_ws)
{
    __shared__ __align__(16) unsigned short ldsA[64][32];   // [m][k] bf16
    __shared__ __align__(16) unsigned short ldsBt[64][32];  // [n][k] bf16

    const int which = blockIdx.z;
    const unsigned short* W = wbf + (size_t)which * DIMC * DIMC;
    unsigned short* dst = qkv_ws + (size_t)which * ((size_t)NBH * SEQT * HD);

    const int bm = blockIdx.x;           // 0..127 : 64-row slab of x
    const int bn = blockIdx.y;           // 0..15  : 64-col slab of W^T
    const int tid  = threadIdx.x;
    const int lane = tid & 31, wave = tid >> 5;
    const int wm = wave & 3, wp = wave >> 2;   // wave -> (m-tile, n-pair)
    const int col = lane & 15, hi = lane >> 4;

    const int fr = tid >> 2, fc = (tid & 3) * 8;   // 16B-chunk per thread

    v8f acc0 = {0,0,0,0,0,0,0,0};
    v8f acc1 = {0,0,0,0,0,0,0,0};

    for (int k0 = 0; k0 < DIMC; k0 += 32) {
        __syncthreads();
        {
            const unsigned short* aS = xbf + (size_t)(bm * 64 + fr) * DIMC + k0 + fc;
            const unsigned short* bS = W   + (size_t)(bn * 64 + fr) * DIMC + k0 + fc;
            unsigned short* aD = &ldsA[0][0]  + tid * 8;
            unsigned short* bD = &ldsBt[0][0] + tid * 8;
#if HAVE_ASYNC_LDS
            ASYNC_B128(aS, aD);                      // DMA 16B, no VGPR trip
            ASYNC_B128(bS, bD);
#else
            #pragma unroll
            for (int q = 0; q < 4; ++q) {
                reinterpret_cast<unsigned int*>(aD)[q] =
                    reinterpret_cast<const unsigned int*>(aS)[q];
                reinterpret_cast<unsigned int*>(bD)[q] =
                    reinterpret_cast<const unsigned int*>(bS)[q];
            }
#endif
            if (k0 + 32 < DIMC) {                    // prefetch next K-slab
                __builtin_prefetch(aS + 32, 0, 1);
                __builtin_prefetch(bS + 32, 0, 1);
            }
        }
        wait_async_lds();
        __syncthreads();

        Frag16 a;
        {
            int row = wm * 16 + col;
            int kbase = hi * 8;
            #pragma unroll
            for (int j = 0; j < 8; ++j) {
                int kk = ((j & 4) ? 16 : 0) + kbase + 2 * (j & 3);
                a.u[j] = *reinterpret_cast<const unsigned int*>(&ldsA[row][kk]);
            }
        }
        Frag16 b0, b1;
        {
            int kbase = hi * 8;
            #pragma unroll
            for (int j = 0; j < 8; ++j) {
                int kk = ((j & 4) ? 16 : 0) + kbase + 2 * (j & 3);
                b0.u[j] = *reinterpret_cast<const unsigned int*>(&ldsBt[wp * 32 + col][kk]);
                b1.u[j] = *reinterpret_cast<const unsigned int*>(&ldsBt[wp * 32 + 16 + col][kk]);
            }
        }
        acc0 = __builtin_amdgcn_wmma_f32_16x16x32_bf16(false, a.v, false, b0.v,
                                                       (short)0, acc0, false, false);
        acc1 = __builtin_amdgcn_wmma_f32_16x16x32_bf16(false, a.v, false, b1.v,
                                                       (short)0, acc1, false, false);
    }

    const int m_base = bm * 64 + wm * 16;
    #pragma unroll
    for (int t2 = 0; t2 < 2; ++t2) {
        v8f acc = t2 ? acc1 : acc0;
        int n_g = bn * 64 + wp * 32 + t2 * 16 + col;
        int h = n_g >> 6, d = n_g & 63;
        #pragma unroll
        for (int r = 0; r < 8; ++r) {
            int m = m_base + r + 8 * hi;
            int bb = m >> 11, t = m & 2047;
            dst[(((size_t)(bb * 16 + h)) * SEQT + t) * HD + d] = f32_to_bf16(acc[r]);
        }
    }
}

// ===========================================================================
// Kernel 2: causal flash attention per (b,h).  Q,K,V bf16 [bh][T][64].
// Block = 256 thr: 8 waves x 16 queries = 128-query slab; key blocks of 64.
// ===========================================================================
__global__ __launch_bounds__(256) void attn_kernel(
    const unsigned short* __restrict__ qkv_ws,
    unsigned short* __restrict__ o_ws)
{
    __shared__ __align__(16) unsigned short ldsK [64][64];   // [key][d]
    __shared__ __align__(16) unsigned short ldsVt[64][64];   // [d][key]
    __shared__ __align__(16) unsigned short ldsP [8][16][64];// per-wave P patch

    const int qblk = blockIdx.x;     // 0..15
    const int bh   = blockIdx.y;     // 0..63
    const int tid  = threadIdx.x;
    const int lane = tid & 31, wave = tid >> 5;
    const int col = lane & 15, hi = lane >> 4;

    const size_t plane = (size_t)NBH * SEQT * HD;
    const unsigned short* Q = qkv_ws + (size_t)bh * SEQT * HD;
    const unsigned short* K = qkv_ws + plane     + (size_t)bh * SEQT * HD;
    const unsigned short* V = qkv_ws + 2 * plane + (size_t)bh * SEQT * HD;

    const int q_start = qblk * 128;
    const int q0 = q_start + wave * 16;

    // ---- preload Q A-fragments (two 32-wide K halves of d) ----------------
    Frag16 qA0, qA1;
    {
        int row = q0 + col;
        int kbase = hi * 8;
        #pragma unroll
        for (int j = 0; j < 8; ++j) {
            int kk0 = ((j & 4) ? 16 : 0) + kbase + 2 * (j & 3);
            qA0.u[j] = *reinterpret_cast<const unsigned int*>(&Q[(size_t)row * HD + kk0]);
            qA1.u[j] = *reinterpret_cast<const unsigned int*>(&Q[(size_t)row * HD + 32 + kk0]);
        }
    }

    float m_run[8], l_run[8];
    #pragma unroll
    for (int r = 0; r < 8; ++r) { m_run[r] = -3.0e38f; l_run[r] = 0.0f; }
    v8f zero = {0,0,0,0,0,0,0,0};
    v8f accO[4];
    #pragma unroll
    for (int dt = 0; dt < 4; ++dt) accO[dt] = zero;

    const int kb_max = (q_start + 127) >> 6;       // inclusive (causal bound)
    for (int kb = 0; kb <= kb_max; ++kb) {
        __syncthreads();
        {
            const unsigned short* Kbase = K + (size_t)kb * 64 * HD;
            const unsigned short* Vbase = V + (size_t)kb * 64 * HD;
            unsigned short* ldsKflat = &ldsK[0][0];
#if HAVE_ASYNC_LDS
            // K tile: straight copy -> async DMA (ASYNCcnt), 2x16B per thread
            #pragma unroll
            for (int i = 0; i < 2; ++i) {
                int idx = tid * 2 + i;             // 0..511 16B chunks
                ASYNC_B128(Kbase + idx * 8, ldsKflat + idx * 8);
            }
#else
            #pragma unroll
            for (int i = 0; i < 8; ++i) {
                int idx = tid * 8 + i;             // 0..2047 dwords
                *reinterpret_cast<unsigned int*>(&ldsKflat[idx * 2]) =
                    reinterpret_cast<const unsigned int*>(Kbase)[idx];
            }
#endif
            // V tile: transpose into [d][key] (needs element scatter)
            #pragma unroll
            for (int i = 0; i < 8; ++i) {
                int idx = tid * 8 + i;
                int key = idx >> 5, dp = idx & 31; // dp = d-pair index
                unsigned int u = reinterpret_cast<const unsigned int*>(Vbase)[idx];
                ldsVt[dp * 2][key]     = (unsigned short)(u & 0xFFFFu);
                ldsVt[dp * 2 + 1][key] = (unsigned short)(u >> 16);
            }
            if (kb < kb_max) {                     // prefetch next key block
                __builtin_prefetch(Kbase + 64 * HD + tid * 16, 0, 1);
                __builtin_prefetch(Vbase + 64 * HD + tid * 16, 0, 1);
            }
        }
        wait_async_lds();
        __syncthreads();

        if (kb * 64 <= q0 + 15) {                  // wave-uniform causal skip
            // ---------------- S = (Q K^T) over this 64-key block -----------
            v8f S[4];
            #pragma unroll
            for (int c = 0; c < 4; ++c) S[c] = zero;
            #pragma unroll
            for (int c = 0; c < 4; ++c) {
                #pragma unroll
                for (int h = 0; h < 2; ++h) {
                    Frag16 b;
                    int kbase = hi * 8;
                    #pragma unroll
                    for (int j = 0; j < 8; ++j)
                        b.u[j] = *reinterpret_cast<const unsigned int*>(
                            &ldsK[c * 16 + col][h * 32 + ((j & 4) ? 16 : 0) + kbase + 2 * (j & 3)]);
                    S[c] = __builtin_amdgcn_wmma_f32_16x16x32_bf16(
                        false, (h ? qA1.v : qA0.v), false, b.v,
                        (short)0, S[c], false, false);
                }
            }
            // ---------------- online softmax (row = 16-lane group) ---------
            #pragma unroll
            for (int r = 0; r < 8; ++r) {
                int qi = q0 + r + 8 * hi;
                float sv[4];
                float mloc = -3.0e38f;
                #pragma unroll
                for (int c = 0; c < 4; ++c) {
                    int kj = kb * 64 + c * 16 + col;
                    float s = S[c][r] * 0.125f;          // 1/sqrt(64)
                    if (kj > qi) s = -3.0e38f;           // causal mask
                    sv[c] = s;
                    mloc = fmaxf(mloc, s);
                }
                #pragma unroll
                for (int mm = 1; mm < 16; mm <<= 1)
                    mloc = fmaxf(mloc, __shfl_xor(mloc, mm, 32));
                float mnew  = fmaxf(m_run[r], mloc);
                float scale = __expf(m_run[r] - mnew);
                float psum = 0.0f;
                #pragma unroll
                for (int c = 0; c < 4; ++c) {
                    float p = __expf(sv[c] - mnew);
                    psum += p;
                    ldsP[wave][r + 8 * hi][c * 16 + col] = f32_to_bf16(p);
                }
                #pragma unroll
                for (int mm = 1; mm < 16; mm <<= 1)
                    psum += __shfl_xor(psum, mm, 32);
                l_run[r] = l_run[r] * scale + psum;
                m_run[r] = mnew;
                #pragma unroll
                for (int dt = 0; dt < 4; ++dt) accO[dt][r] *= scale;
            }
            __threadfence_block();   // order ldsP stores vs re-typed loads

            // ---------------- O += P V -------------------------------------
            Frag16 pA0, pA1;
            {
                int kbase = hi * 8;
                #pragma unroll
                for (int j = 0; j < 8; ++j) {
                    int kk0 = ((j & 4) ? 16 : 0) + kbase + 2 * (j & 3);
                    pA0.u[j] = *reinterpret_cast<const unsigned int*>(&ldsP[wave][col][kk0]);
                    pA1.u[j] = *reinterpret_cast<const unsigned int*>(&ldsP[wave][col][32 + kk0]);
                }
            }
            #pragma unroll
            for (int dt = 0; dt < 4; ++dt) {
                #pragma unroll
                for (int h = 0; h < 2; ++h) {
                    Frag16 b;
                    int kbase = hi * 8;
                    #pragma unroll
                    for (int j = 0; j < 8; ++j)
                        b.u[j] = *reinterpret_cast<const unsigned int*>(
                            &ldsVt[dt * 16 + col][h * 32 + ((j & 4) ? 16 : 0) + kbase + 2 * (j & 3)]);
                    accO[dt] = __builtin_amdgcn_wmma_f32_16x16x32_bf16(
                        false, (h ? pA1.v : pA0.v), false, b.v,
                        (short)0, accO[dt], false, false);
                }
            }
        }
    }

    // ---- normalize + write O as bf16 in [b][t][h*64+d] (ready for Wo GEMM)
    const int b = bh >> 4, h = bh & 15;
    #pragma unroll
    for (int dt = 0; dt < 4; ++dt) {
        int d = dt * 16 + col;
        #pragma unroll
        for (int r = 0; r < 8; ++r) {
            int t = q0 + r + 8 * hi;
            float o = accO[dt][r] / l_run[r];
            o_ws[((size_t)(b * SEQT + t)) * DIMC + h * HD + d] = f32_to_bf16(o);
        }
    }
}

// ===========================================================================
// Kernel 3: output projection  out = O @ Wo^T + bo   (fp32 result)
// ===========================================================================
__global__ __launch_bounds__(256) void out_proj_kernel(
    const unsigned short* __restrict__ O,
    const unsigned short* __restrict__ Wobf, const float* __restrict__ bo,
    float* __restrict__ out)
{
    __shared__ __align__(16) unsigned short ldsA[64][32];
    __shared__ __align__(16) unsigned short ldsBt[64][32];

    const int bm = blockIdx.x;           // 0..127
    const int bn = blockIdx.y;           // 0..15
    const int tid  = threadIdx.x;
    const int lane = tid & 31, wave = tid >> 5;
    const int wm = wave & 3, wp = wave >> 2;
    const int col = lane & 15, hi = lane >> 4;

    const int fr = tid >> 2, fc = (tid & 3) * 8;   // 16B-chunk per thread

    v8f acc0 = {0,0,0,0,0,0,0,0};
    v8f acc1 = {0,0,0,0,0,0,0,0};

    for (int k0 = 0; k0 < DIMC; k0 += 32) {
        __syncthreads();
        {
            const unsigned short* aS = O    + (size_t)(bm * 64 + fr) * DIMC + k0 + fc;
            const unsigned short* bS = Wobf + (size_t)(bn * 64 + fr) * DIMC + k0 + fc;
            unsigned short* aD = &ldsA[0][0]  + tid * 8;
            unsigned short* bD = &ldsBt[0][0] + tid * 8;
#if HAVE_ASYNC_LDS
            ASYNC_B128(aS, aD);
            ASYNC_B128(bS, bD);
#else
            #pragma unroll
            for (int q = 0; q < 4; ++q) {
                reinterpret_cast<unsigned int*>(aD)[q] =
                    reinterpret_cast<const unsigned int*>(aS)[q];
                reinterpret_cast<unsigned int*>(bD)[q] =
                    reinterpret_cast<const unsigned int*>(bS)[q];
            }
#endif
            if (k0 + 32 < DIMC) {
                __builtin_prefetch(aS + 32, 0, 1);
                __builtin_prefetch(bS + 32, 0, 1);
            }
        }
        wait_async_lds();
        __syncthreads();

        Frag16 a;
        {
            int row = wm * 16 + col;
            int kbase = hi * 8;
            #pragma unroll
            for (int j = 0; j < 8; ++j) {
                int kk = ((j & 4) ? 16 : 0) + kbase + 2 * (j & 3);
                a.u[j] = *reinterpret_cast<const unsigned int*>(&ldsA[row][kk]);
            }
        }
        Frag16 b0, b1;
        {
            int kbase = hi * 8;
            #pragma unroll
            for (int j = 0; j < 8; ++j) {
                int kk = ((j & 4) ? 16 : 0) + kbase + 2 * (j & 3);
                b0.u[j] = *reinterpret_cast<const unsigned int*>(&ldsBt[wp * 32 + col][kk]);
                b1.u[j] = *reinterpret_cast<const unsigned int*>(&ldsBt[wp * 32 + 16 + col][kk]);
            }
        }
        acc0 = __builtin_amdgcn_wmma_f32_16x16x32_bf16(false, a.v, false, b0.v,
                                                       (short)0, acc0, false, false);
        acc1 = __builtin_amdgcn_wmma_f32_16x16x32_bf16(false, a.v, false, b1.v,
                                                       (short)0, acc1, false, false);
    }

    const int m_base = bm * 64 + wm * 16;
    #pragma unroll
    for (int t2 = 0; t2 < 2; ++t2) {
        v8f acc = t2 ? acc1 : acc0;
        int n_g = bn * 64 + wp * 32 + t2 * 16 + col;
        float bias = bo[n_g];
        #pragma unroll
        for (int r = 0; r < 8; ++r) {
            int m = m_base + r + 8 * hi;
            out[(size_t)m * DIMC + n_g] = acc[r] + bias;
        }
    }
}

// ===========================================================================
extern "C" void kernel_launch(void* const* d_in, const int* in_sizes, int n_in,
                              void* d_out, int out_size, void* d_ws, size_t ws_size,
                              hipStream_t stream) {
    const float* x  = (const float*)d_in[0];
    const float* Wq = (const float*)d_in[1];
    const float* Wk = (const float*)d_in[2];
    const float* Wv = (const float*)d_in[3];
    const float* Wo = (const float*)d_in[4];
    const float* bo = (const float*)d_in[5];
    float* out = (float*)d_out;

    // Workspace (ushort units):
    //   [0, 24M)   : Q|K|V bf16, [bh][T][64] each               (48 MB)
    //   [24M, 32M) : xbf (pre-pass) then O (attention out) -- aliased (16 MB)
    //   [32M, 36M) : Wq|Wk|Wv|Wo bf16, [n][k]                   ( 8 MB)
    unsigned short* qkv = (unsigned short*)d_ws;
    unsigned short* xo  = qkv + (size_t)3 * NBH * SEQT * HD;     // x / O alias
    unsigned short* wbf = xo  + (size_t)SEQT * 4 * DIMC;

    convert_kernel <<<dim3(4096, 5), 256, 0, stream>>>(x, Wq, Wk, Wv, Wo, xo, wbf);
    qkv_proj_kernel<<<dim3(128, 16, 3), 256, 0, stream>>>(xo, wbf, qkv);
    attn_kernel    <<<dim3(16, 64),     256, 0, stream>>>(qkv, xo);
    out_proj_kernel<<<dim3(128, 16),    256, 0, stream>>>(
        xo, wbf + (size_t)3 * DIMC * DIMC, bo, out);
}
